// FactorizedAttention_74474732912893
// MI455X (gfx1250) — matrix operand order repaired and checked by reading, not compiled, not run on hardware
//
#include <hip/hip_runtime.h>
#include <hip/hip_bf16.h>

// ---------------------------------------------------------------------------
// FactorizedAttention (CoaT) for MI455X / gfx1250.
// All GEMMs run on v_wmma_f32_16x16x32_bf16 (wave32). Large GEMMs use 4x2
// fragment register blocking (64x32 macro-tile per wave): ~112 live VGPRs of
// data, which fits the allocator's budget WITHOUT accumulator spills (4x4
// blocking spilled to scratch inside the K-loop in the previous round).
// B operands are stored transposed (N-major over K) so a B fragment is one
// contiguous 32-byte load per lane. fp32 accumulation throughout.
// ---------------------------------------------------------------------------

typedef __attribute__((ext_vector_type(16))) __bf16 v16bf;
typedef __attribute__((ext_vector_type(8)))  float  v8f;

#define Bsz   16
#define Ntok  1025
#define Cch   768
#define NH    12
#define DH    64
#define HW    32
#define NPAD  1056   // 1025 padded up to a multiple of 32 (K dim of kTv GEMM)
#define MROWS (Bsz * Ntok)   // 16400, divisible by 16
#define MBLK  ((MROWS + 63) / 64)  // 257 64-row M blocks (overshoot guarded)

union FragU {
    v16bf v;
    uint4 u[2];
};

// A fragment (16x32 bf16, row-major A, lda in elements).
// Lane L: M = L&15. Lanes<16: K 0..15; lanes>=16: K 16..31 (contiguous).
// -> two b128 loads.
__device__ __forceinline__ v16bf load_afrag(const __hip_bfloat16* A, int row,
                                            int lda, int kbase, int lane) {
    const __hip_bfloat16* s = A + (size_t)row * lda + kbase + ((lane & 16) ? 16 : 0);
    FragU f;
    f.u[0] = *(const uint4*)(s);
    f.u[1] = *(const uint4*)(s + 8);
    return f.v;
}

// B fragment (32x16 bf16) from B stored TRANSPOSED: Bt is (Ncols x K) row-major.
// Lane L: N = L&15, K = ((L>=16)?16:0) + 2v + {0,1}  -> 16 contiguous K values.
__device__ __forceinline__ v16bf load_bfrag(const __hip_bfloat16* Bt, int col,
                                            int ldb, int kbase, int lane) {
    const __hip_bfloat16* s = Bt + (size_t)col * ldb + kbase + ((lane & 16) ? 16 : 0);
    FragU f;
    f.u[0] = *(const uint4*)(s);
    f.u[1] = *(const uint4*)(s + 8);
    return f.v;
}

// Single 16x16 tile (used by the small per-head kTv GEMM).
__device__ __forceinline__ void wmma_tile(const __hip_bfloat16* A, int lda,
                                          const __hip_bfloat16* Bt, int ldb,
                                          int mbase, int nbase, int K,
                                          int lane, v8f& acc) {
    const int arow = mbase + (lane & 15);
    const int bcol = nbase + (lane & 15);
    for (int k = 0; k < K; k += 32) {
        if ((k & 127) == 0) {  // emits global_prefetch_b8 (speculative)
            __builtin_prefetch((const void*)(A + (size_t)arow * lda + k + 128), 0, 1);
            __builtin_prefetch((const void*)(Bt + (size_t)bcol * ldb + k + 128), 0, 1);
        }
        v16bf a = load_afrag(A, arow, lda, k, lane);
        v16bf b = load_bfrag(Bt, bcol, ldb, k, lane);
        acc = __builtin_amdgcn_wmma_f32_16x16x32_bf16(
            false, a, false, b, (short)0, acc, false, false);
    }
}

// 4x2-fragment register-blocked macro-tile: C(64x32) per wave.
// 8 v8f accumulators (64 VGPRs) + 4 A frags (32) + 2 B frags (16): no spills.
__device__ __forceinline__ void wmma_blocked4x2(const __hip_bfloat16* A, int lda,
                                                const __hip_bfloat16* Bt, int ldb,
                                                int mbase, int nbase, int K,
                                                int lane, v8f acc[4][2]) {
    const int ar = mbase + (lane & 15);
    const int bc = nbase + (lane & 15);
    const int aoff = (lane & 16) ? 16 : 0;
    const int boff = (lane & 16) ? 16 : 0;
    for (int k = 0; k < K; k += 32) {
        v16bf a[4], b[2];
#pragma unroll
        for (int i = 0; i < 4; i++) {
            const __hip_bfloat16* s = A + (size_t)(ar + 16 * i) * lda + k + aoff;
            FragU f;
            f.u[0] = *(const uint4*)(s);
            f.u[1] = *(const uint4*)(s + 8);
            a[i] = f.v;
        }
#pragma unroll
        for (int j = 0; j < 2; j++) {
            const __hip_bfloat16* s = Bt + (size_t)(bc + 16 * j) * ldb + k + boff;
            FragU f;
            f.u[0] = *(const uint4*)(s);
            f.u[1] = *(const uint4*)(s + 8);
            b[j] = f.v;
        }
#pragma unroll
        for (int i = 0; i < 4; i++)
#pragma unroll
            for (int j = 0; j < 2; j++)
                acc[i][j] = __builtin_amdgcn_wmma_f32_16x16x32_bf16(
                    false, a[i], false, b[j], (short)0, acc[i][j], false, false);
    }
}

// ---------------------------------------------------------------------------
// 1) CPE: 3x3 depthwise conv + residual on image tokens; cls passthrough.
// ---------------------------------------------------------------------------
__global__ void cpe_kernel(const float* __restrict__ in,
                           const float* __restrict__ cpe_w,
                           const float* __restrict__ cpe_b,
                           __hip_bfloat16* __restrict__ peb) {
    long idx = (long)blockIdx.x * blockDim.x + threadIdx.x;
    if (idx >= (long)Bsz * Ntok * Cch) return;
    int c = idx % Cch;
    long t = idx / Cch;
    int n = t % Ntok;
    int b = t / Ntok;
    float val;
    if (n == 0) {
        val = in[idx];
    } else {
        int p = n - 1, h = p / HW, w = p % HW;
        float acc = cpe_b[c];
#pragma unroll
        for (int i = -1; i <= 1; i++)
#pragma unroll
            for (int j = -1; j <= 1; j++) {
                int hh = h + i, ww = w + j;
                if (hh >= 0 && hh < HW && ww >= 0 && ww < HW)
                    acc += in[(((long)b * Ntok + 1 + hh * HW + ww) * Cch) + c] *
                           cpe_w[((i + 1) * 3 + (j + 1)) * Cch + c];
            }
        val = acc + in[idx];
    }
    peb[idx] = __float2bfloat16(val);
}

// ---------------------------------------------------------------------------
// 2) Transpose + convert weights to bf16: wqkvT (2304x768), wprojT (768x768)
// ---------------------------------------------------------------------------
__global__ void wcvt_kernel(const float* __restrict__ w_qkv,
                            const float* __restrict__ w_proj,
                            __hip_bfloat16* __restrict__ wqkvT,
                            __hip_bfloat16* __restrict__ wprojT) {
    int idx = blockIdx.x * blockDim.x + threadIdx.x;
    const int n1 = 3 * Cch * Cch;
    const int n2 = Cch * Cch;
    if (idx < n1) {
        int nn = idx / Cch, k = idx % Cch;
        wqkvT[idx] = __float2bfloat16(w_qkv[(long)k * (3 * Cch) + nn]);
    } else if (idx < n1 + n2) {
        int r = idx - n1;
        int nn = r / Cch, k = r % Cch;
        wprojT[r] = __float2bfloat16(w_proj[(long)k * Cch + nn]);
    }
}

// ---------------------------------------------------------------------------
// 3) Zero padding rows/cols (tokens 1025..1055) for q (fp32) and vT (bf16)
// ---------------------------------------------------------------------------
__global__ void padzero_kernel(float* __restrict__ qf32,
                               __hip_bfloat16* __restrict__ vTb) {
    int idx = blockIdx.x * blockDim.x + threadIdx.x;
    const int total = Bsz * NH * DH * (NPAD - Ntok);
    if (idx >= total) return;
    {   // q pad rows
        int d = idx & 63;
        int t = idx >> 6;
        int r = t % (NPAD - Ntok);
        int bh = t / (NPAD - Ntok);
        qf32[((long)bh * NPAD + Ntok + r) * DH + d] = 0.0f;
    }
    {   // vT pad cols
        int cr = idx % (NPAD - Ntok);
        int t = idx / (NPAD - Ntok);
        int d = t & 63;
        int bh = t >> 6;
        vTb[((long)bh * DH + d) * NPAD + Ntok + cr] = __float2bfloat16(0.0f);
    }
}

// ---------------------------------------------------------------------------
// 4) QKV GEMM: (16400 x 768) @ (768 x 2304), 64x32 macro-tile per wave,
//    scatter into q/kT/v/vT layouts.
// ---------------------------------------------------------------------------
__global__ void __launch_bounds__(128, 1)
qkv_gemm_kernel(const __hip_bfloat16* __restrict__ peb,
                const __hip_bfloat16* __restrict__ wqkvT,
                float* __restrict__ qf32,
                float* __restrict__ kTf,
                float* __restrict__ vf32,
                __hip_bfloat16* __restrict__ vTb) {
    int wave = (blockIdx.x * blockDim.x + threadIdx.x) >> 5;
    int lane = threadIdx.x & 31;
    const int NBLK = (3 * Cch) / 32;  // 72
    if (wave >= MBLK * NBLK) return;
    int mb = wave / NBLK, nb = wave % NBLK;

    v8f acc[4][2];
    v8f z = {};
#pragma unroll
    for (int i = 0; i < 4; i++)
#pragma unroll
        for (int j = 0; j < 2; j++) acc[i][j] = z;

    wmma_blocked4x2(peb, Cch, wqkvT, Cch, mb * 64, nb * 32, Cch, lane, acc);

#pragma unroll
    for (int j = 0; j < 2; j++) {
        int col = nb * 32 + j * 16 + (lane & 15);
        int which = col / Cch;           // uniform over the 16-col group
        int c = col % Cch;
        int h = c >> 6, d = c & 63;
#pragma unroll
        for (int i = 0; i < 4; i++) {
#pragma unroll
            for (int vi = 0; vi < 8; vi++) {
                int row = mb * 64 + i * 16 + vi + ((lane & 16) ? 8 : 0);
                if (row < MROWS) {
                    int b = row / Ntok, ntk = row % Ntok;
                    int bh = b * NH + h;
                    float val = acc[i][j][vi];
                    if (which == 0) {
                        qf32[((long)bh * NPAD + ntk) * DH + d] = val;
                    } else if (which == 1) {
                        kTf[((long)bh * DH + d) * NPAD + ntk] = val;
                    } else {
                        vf32[((long)bh * Ntok + ntk) * DH + d] = val;
                        vTb[((long)bh * DH + d) * NPAD + ntk] = __float2bfloat16(val);
                    }
                }
            }
        }
    }
}

// ---------------------------------------------------------------------------
// 5) Softmax over tokens (axis=N) per (b,h,k-channel): one wave32 per row
// ---------------------------------------------------------------------------
__global__ void softmax_kernel(const float* __restrict__ kTf,
                               __hip_bfloat16* __restrict__ ksoftb) {
    int wave = (blockIdx.x * blockDim.x + threadIdx.x) >> 5;
    int lane = threadIdx.x & 31;
    if (wave >= Bsz * NH * DH) return;
    const float* src = kTf + (long)wave * NPAD;
    __hip_bfloat16* dst = ksoftb + (long)wave * NPAD;

    float mx = -3.402823466e+38f;
    for (int i = lane; i < Ntok; i += 32) mx = fmaxf(mx, src[i]);
#pragma unroll
    for (int o = 16; o; o >>= 1) mx = fmaxf(mx, __shfl_xor(mx, o, 32));
    float sum = 0.0f;
    for (int i = lane; i < Ntok; i += 32) sum += __expf(src[i] - mx);
#pragma unroll
    for (int o = 16; o; o >>= 1) sum += __shfl_xor(sum, o, 32);
    float inv = 1.0f / sum;
    for (int i = lane; i < NPAD; i += 32)
        dst[i] = __float2bfloat16(i < Ntok ? __expf(src[i] - mx) * inv : 0.0f);
}

// ---------------------------------------------------------------------------
// 6) kTv GEMM per head: (64 x 1056) @ (1056 x 64) -> store transposed bf16.
//    Kept at 16x16/wave for parallelism (3072 waves); traffic is negligible.
// ---------------------------------------------------------------------------
__global__ void ktv_gemm_kernel(const __hip_bfloat16* __restrict__ ksoftb,
                                const __hip_bfloat16* __restrict__ vTb,
                                __hip_bfloat16* __restrict__ kTvTb) {
    int wave = (blockIdx.x * blockDim.x + threadIdx.x) >> 5;
    int lane = threadIdx.x & 31;
    if (wave >= Bsz * NH * 16) return;
    int bh = wave >> 4;
    int tile = wave & 15;
    int mt = tile >> 2, nt = tile & 3;
    const __hip_bfloat16* A  = ksoftb + (long)bh * DH * NPAD;
    const __hip_bfloat16* Bt = vTb + (long)bh * DH * NPAD;

    v8f acc = {};
    wmma_tile(A, NPAD, Bt, NPAD, mt * 16, nt * 16, NPAD, lane, acc);

    int nc = nt * 16 + (lane & 15);
#pragma unroll
    for (int vi = 0; vi < 8; vi++) {
        int mr = mt * 16 + vi + ((lane & 16) ? 8 : 0);
        kTvTb[((long)bh * DH + nc) * DH + mr] = __float2bfloat16(acc[vi]);
    }
}

// ---------------------------------------------------------------------------
// 7) q fp32 -> bf16 (pads already zeroed, so bf16 pads are zero too)
// ---------------------------------------------------------------------------
__global__ void qcvt_kernel(const float* __restrict__ qf32,
                            __hip_bfloat16* __restrict__ qb) {
    long idx = (long)blockIdx.x * blockDim.x + threadIdx.x;
    if (idx >= (long)Bsz * NH * NPAD * DH) return;
    qb[idx] = __float2bfloat16(qf32[idx]);
}

// ---------------------------------------------------------------------------
// 8) factor_att GEMM per head (1056 x 64) @ (64 x 64), 64x32 macro-tile per
//    wave, with exact reference CRPE-scramble epilogue.
// ---------------------------------------------------------------------------
__global__ void __launch_bounds__(128, 1)
factor_gemm_kernel(const __hip_bfloat16* __restrict__ qb,
                   const __hip_bfloat16* __restrict__ kTvTb,
                   const float* __restrict__ qf32,
                   const float* __restrict__ vf32,
                   const float* __restrict__ crpe_w,
                   const float* __restrict__ crpe_b,
                   __hip_bfloat16* __restrict__ attnb) {
    int wave = (blockIdx.x * blockDim.x + threadIdx.x) >> 5;
    int lane = threadIdx.x & 31;
    const int MB = (NPAD + 63) / 64;   // 17 (overshoot rows guarded on store)
    if (wave >= Bsz * NH * MB * 2) return;
    int bh = wave / (MB * 2);
    int r = wave % (MB * 2);
    int mblk = r >> 1;
    int nblk = r & 1;
    const __hip_bfloat16* A  = qb + (long)bh * NPAD * DH;
    const __hip_bfloat16* Bt = kTvTb + (long)bh * DH * DH;

    v8f acc[4][2];
    v8f z = {};
#pragma unroll
    for (int i = 0; i < 4; i++)
#pragma unroll
        for (int j = 0; j < 2; j++) acc[i][j] = z;

    wmma_blocked4x2(A, DH, Bt, DH, mblk * 64, nblk * 32, DH, lane, acc);

    int h = bh % NH, b = bh / NH;
    const float scale = 0.125f;  // dh^-0.5
#pragma unroll
    for (int j = 0; j < 2; j++) {
        int d = nblk * 32 + j * 16 + (lane & 15);
#pragma unroll
        for (int i = 0; i < 4; i++) {
#pragma unroll
            for (int vi = 0; vi < 8; vi++) {
                int ntk = mblk * 64 + i * 16 + vi + ((lane & 16) ? 8 : 0);
                if (ntk < Ntok) {
                    float f = scale * acc[i][j][vi];
                    if (ntk >= 1) {
                        // exact reference CRPE index scramble:
                        // (nh,1024,dh)->(768,32,32)->NHWC 1x1 dwconv->(h,w,768)->(nh,1024,dh)
                        int n2 = ntk - 1;
                        int flat4 = h * 65536 + n2 * 64 + d;
                        int cch = flat4 % Cch;
                        int w2 = (flat4 / Cch) % HW;
                        int h2 = flat4 / (Cch * HW);
                        int flat3 = cch * 1024 + h2 * HW + w2;
                        int head1 = flat3 >> 16;
                        int n1 = (flat3 >> 6) & 1023;
                        int d1 = flat3 & 63;
                        float qv = qf32[((long)bh * NPAD + ntk) * DH + d];
                        float vv = vf32[((long)(b * NH + head1) * Ntok + 1 + n1) * DH + d1];
                        f += qv * (vv * crpe_w[cch] + crpe_b[cch]);
                    }
                    attnb[(((long)b * Ntok + ntk) * Cch) + h * DH + d] = __float2bfloat16(f);
                }
            }
        }
    }
}

// ---------------------------------------------------------------------------
// 9) Projection GEMM: (16400 x 768) @ (768 x 768) + bias -> fp32 output,
//    64x32 macro-tile per wave.
// ---------------------------------------------------------------------------
__global__ void __launch_bounds__(128, 1)
proj_gemm_kernel(const __hip_bfloat16* __restrict__ attnb,
                 const __hip_bfloat16* __restrict__ wprojT,
                 const float* __restrict__ b_proj,
                 float* __restrict__ out) {
    int wave = (blockIdx.x * blockDim.x + threadIdx.x) >> 5;
    int lane = threadIdx.x & 31;
    const int NBLK = Cch / 32;  // 24
    if (wave >= MBLK * NBLK) return;
    int mb = wave / NBLK, nb = wave % NBLK;

    v8f acc[4][2];
    v8f z = {};
#pragma unroll
    for (int i = 0; i < 4; i++)
#pragma unroll
        for (int j = 0; j < 2; j++) acc[i][j] = z;

    wmma_blocked4x2(attnb, Cch, wprojT, Cch, mb * 64, nb * 32, Cch, lane, acc);

#pragma unroll
    for (int j = 0; j < 2; j++) {
        int col = nb * 32 + j * 16 + (lane & 15);
        float bias = b_proj[col];
#pragma unroll
        for (int i = 0; i < 4; i++) {
#pragma unroll
            for (int vi = 0; vi < 8; vi++) {
                int row = mb * 64 + i * 16 + vi + ((lane & 16) ? 8 : 0);
                if (row < MROWS)
                    out[(long)row * Cch + col] = acc[i][j][vi] + bias;
            }
        }
    }
}

// ---------------------------------------------------------------------------
extern "C" void kernel_launch(void* const* d_in, const int* in_sizes, int n_in,
                              void* d_out, int out_size, void* d_ws, size_t ws_size,
                              hipStream_t stream) {
    const float* inputs = (const float*)d_in[0];
    const float* cpe_w  = (const float*)d_in[1];
    const float* cpe_b  = (const float*)d_in[2];
    const float* w_qkv  = (const float*)d_in[3];
    const float* crpe_w = (const float*)d_in[4];
    const float* crpe_b = (const float*)d_in[5];
    const float* w_proj = (const float*)d_in[6];
    const float* b_proj = (const float*)d_in[7];
    float* out = (float*)d_out;

    // Workspace layout (all sub-buffers 256B-aligned; ~289 MB + overread pad)
    char* p = (char*)d_ws;
    __hip_bfloat16* peb    = (__hip_bfloat16*)p; p += (size_t)MROWS * Cch * 2;
    __hip_bfloat16* wqkvT  = (__hip_bfloat16*)p; p += (size_t)3 * Cch * Cch * 2;
    __hip_bfloat16* wprojT = (__hip_bfloat16*)p; p += (size_t)Cch * Cch * 2;
    float*          qf32   = (float*)p;          p += (size_t)Bsz * NH * NPAD * DH * 4;
    float*          kTf    = (float*)p;          p += (size_t)Bsz * NH * DH * NPAD * 4;
    float*          vf32   = (float*)p;          p += (size_t)Bsz * NH * Ntok * DH * 4;
    __hip_bfloat16* vTb    = (__hip_bfloat16*)p; p += (size_t)Bsz * NH * DH * NPAD * 2;
    __hip_bfloat16* ksoftb = (__hip_bfloat16*)p; p += (size_t)Bsz * NH * DH * NPAD * 2;
    __hip_bfloat16* kTvTb  = (__hip_bfloat16*)p; p += (size_t)Bsz * NH * DH * DH * 2;
    __hip_bfloat16* qb     = (__hip_bfloat16*)p; p += (size_t)Bsz * NH * NPAD * DH * 2;
    __hip_bfloat16* attnb  = (__hip_bfloat16*)p; p += (size_t)MROWS * Cch * 2;
    p += 131072;  // pad so 64-row M-block overreads of attnb stay in-bounds

    // 1) CPE + residual -> pe (bf16)
    cpe_kernel<<<((long)MROWS * Cch + 255) / 256, 256, 0, stream>>>(inputs, cpe_w, cpe_b, peb);
    // 2) weights -> transposed bf16
    wcvt_kernel<<<((3 * Cch * Cch + Cch * Cch) + 255) / 256, 256, 0, stream>>>(
        w_qkv, w_proj, wqkvT, wprojT);
    // 3) zero token pads in q / vT
    padzero_kernel<<<(Bsz * NH * DH * (NPAD - Ntok) + 255) / 256, 256, 0, stream>>>(qf32, vTb);
    // 4) QKV GEMM (257x72 = 18504 macro-tile waves, 4 waves/block)
    qkv_gemm_kernel<<<(MBLK * 72 + 3) / 4, 128, 0, stream>>>(
        peb, wqkvT, qf32, kTf, vf32, vTb);
    // 5) softmax over tokens (wave per (b,h,d) row)
    softmax_kernel<<<Bsz * NH * DH / 4, 128, 0, stream>>>(kTf, ksoftb);
    // 6) kTv per-head GEMM
    ktv_gemm_kernel<<<Bsz * NH * 16 / 4, 128, 0, stream>>>(ksoftb, vTb, kTvTb);
    // 7) q -> bf16
    qcvt_kernel<<<((long)Bsz * NH * NPAD * DH + 255) / 256, 256, 0, stream>>>(qf32, qb);
    // 8) factor_att GEMM + CRPE epilogue -> attn bf16 (192*17*2 = 6528 waves)
    factor_gemm_kernel<<<(Bsz * NH * 17 * 2 + 3) / 4, 128, 0, stream>>>(
        qb, kTvTb, qf32, vf32, crpe_w, crpe_b, attnb);
    // 9) projection GEMM + bias -> fp32 out (257*24 = 6168 waves)
    proj_gemm_kernel<<<(MBLK * 24 + 3) / 4, 128, 0, stream>>>(
        attnb, wprojT, b_proj, out);
    (void)in_sizes; (void)n_in; (void)out_size; (void)ws_size;
}
